// MultiHeadMamba6mer_28741921145120
// MI455X (gfx1250) — matrix operand order repaired and checked by reading, not compiled, not run on hardware
//
#include <hip/hip_runtime.h>
#include <hip/hip_bf16.h>

// ---------------------------------------------------------------- constants
#define D_MODEL   384
#define D_INNER   768
#define D_STATE   64
#define D_CONV    4
#define HEADDIM   48
#define NHEADS    16
#define NLAYERS   2
#define BB        32
#define LL        512
#define D_IN_PROJ 1680           // 2*768 + 2*64 + 16
#define CONV_DIM  896            // 768 + 2*64
#define ROWS      (BB*LL)        // 16384
#define N_ORDER   60
#define N_FAMILY  427
#define N_GENUS   14216
#define N_SPECIES 23964
#define N_OUT     (N_ORDER+N_FAMILY+N_GENUS+N_SPECIES) // 38667

typedef __bf16 bf16_t;
typedef __attribute__((ext_vector_type(16))) __bf16 v16bf;
typedef __attribute__((ext_vector_type(8)))  __bf16 v8bf;
typedef __attribute__((ext_vector_type(8)))  float  v8f;

__device__ __forceinline__ bf16_t f2bf(float f) {
    union { float f; unsigned u; } x; x.f = f;
    unsigned r = (x.u + 0x7FFFu + ((x.u >> 16) & 1u)) >> 16;
    union { unsigned short s; bf16_t b; } y; y.s = (unsigned short)r;
    return y.b;
}

__device__ __forceinline__ float silu_f(float x) {
    return x / (1.f + __expf(-x));
}

// ------------------------------------------------------------ embedding gather
__global__ void embed_kernel(const int* __restrict__ tok,
                             const float* __restrict__ emb,
                             float* __restrict__ resid) {
    size_t idx = (size_t)blockIdx.x * blockDim.x + threadIdx.x;
    if (idx >= (size_t)ROWS * D_MODEL) return;
    int d = (int)(idx % D_MODEL);
    size_t bl = idx / D_MODEL;
    resid[idx] = emb[(size_t)tok[bl] * D_MODEL + d];
}

// ------------------------------------------------------------ weight fp32 -> bf16^T
// Wt[n*K + k] = bf16(W[k*N + n])   (coalesced reads over n)
__global__ void convert_wT_kernel(const float* __restrict__ W,
                                  bf16_t* __restrict__ Wt, int K, int N) {
    size_t idx = (size_t)blockIdx.x * blockDim.x + threadIdx.x;
    if (idx >= (size_t)K * N) return;
    int n = (int)(idx % N);
    int k = (int)(idx / N);
    Wt[(size_t)n * K + k] = f2bf(W[idx]);
}

// ------------------------------------------------------------ layernorm (D=384)
// 128 threads/block, one row/block, 3 cols/thread. Writes bf16 and/or fp32.
__global__ void layernorm_kernel(const float* __restrict__ x,
                                 const float* __restrict__ w,
                                 const float* __restrict__ b,
                                 bf16_t* __restrict__ out_bf,
                                 float* __restrict__ out_f) {
    int row = blockIdx.x, t = threadIdx.x;
    __shared__ float rs[128], rq[128];
    float v[3]; float s = 0.f, q = 0.f;
    size_t base = (size_t)row * D_MODEL;
#pragma unroll
    for (int j = 0; j < 3; j++) {
        int c = t + j * 128;
        float xv = x[base + c];
        v[j] = xv; s += xv; q += xv * xv;
    }
    rs[t] = s; rq[t] = q; __syncthreads();
    for (int st = 64; st > 0; st >>= 1) {
        if (t < st) { rs[t] += rs[t + st]; rq[t] += rq[t + st]; }
        __syncthreads();
    }
    float mu = rs[0] * (1.f / D_MODEL);
    float var = rq[0] * (1.f / D_MODEL) - mu * mu;
    float sc = rsqrtf(var + 1e-5f);
#pragma unroll
    for (int j = 0; j < 3; j++) {
        int c = t + j * 128;
        float o = (v[j] - mu) * sc * w[c] + b[c];
        if (out_bf) out_bf[base + c] = f2bf(o);
        if (out_f)  out_f[base + c]  = o;
    }
}

// ------------------------------------------------------------ WMMA bf16 GEMM
// C[M,N] = A[M,K](bf16,row-major) @ Bt[N,K](bf16) (+ optional Cin residual)
// Each wave computes a 64x16 strip (4 M-tiles) so the B (weight) fragment is
// reused 4x per k-step: per 32-K step, 1 B load + 4 A loads feed 4 WMMAs.
// M must be a multiple of 64 (16384 = 256*64). 8 waves/block.
__global__ __launch_bounds__(256)
void wmma_gemm_bf16(const bf16_t* __restrict__ A,
                    const bf16_t* __restrict__ Bt,
                    const float* __restrict__ Cin,
                    float* __restrict__ Cout,
                    int N, int K, int tilesN, int totalGroups) {
    int wave = threadIdx.x >> 5;
    int lane = threadIdx.x & 31;
    int grp = blockIdx.x * 8 + wave;
    if (grp >= totalGroups) return;               // wave-uniform exit
    int tg = grp / tilesN;                        // group of 4 M-tiles
    int tn = grp - tg * tilesN;
    int g  = lane >> 4;                           // lane group 0/1
    int mn = lane & 15;

    // A fragment rows: tg*64 + s*16 + mn; element e<8 -> K=k0+8g+e ; e>=8 -> +16
    const bf16_t* Abase = A  + (size_t)(tg * 64 + mn) * K + 8 * g;
    // B fragment: col = tn*16+mn; element e<8 -> K=k0+8g+e ; e>=8 -> +16
    const bf16_t* Bbase = Bt + (size_t)(tn * 16 + mn) * K + 8 * g;
    const size_t Astride = (size_t)16 * K;

    v8f acc[4];
#pragma unroll
    for (int s = 0; s < 4; s++) acc[s] = (v8f){};

    for (int k0 = 0; k0 < K; k0 += 32) {
        v8bf blo = *(const v8bf*)(Bbase + k0);
        v8bf bhi = *(const v8bf*)(Bbase + k0 + 16);
        v16bf bf;
#pragma unroll
        for (int i = 0; i < 8; i++) { bf[i] = blo[i]; bf[i + 8] = bhi[i]; }
#pragma unroll
        for (int s = 0; s < 4; s++) {
            const bf16_t* ap = Abase + (size_t)s * Astride + k0;
            v8bf alo = *(const v8bf*)(ap);
            v8bf ahi = *(const v8bf*)(ap + 16);
            v16bf a;
#pragma unroll
            for (int i = 0; i < 8; i++) { a[i] = alo[i]; a[i + 8] = ahi[i]; }
            acc[s] = __builtin_amdgcn_wmma_f32_16x16x32_bf16(
                false, a, false, bf, (short)0, acc[s], false, false);
        }
    }
    // D layout per tile: VGPR r -> row +8g + r, col tn*16 + mn
#pragma unroll
    for (int s = 0; s < 4; s++) {
        size_t base = (size_t)(tg * 64 + s * 16 + 8 * g) * N + tn * 16 + mn;
#pragma unroll
        for (int r = 0; r < 8; r++) {
            size_t off = base + (size_t)r * N;
            float v = acc[s][r];
            if (Cin) v += Cin[off];
            Cout[off] = v;
        }
    }
}

// ------------------------------------------------------------ depthwise conv + SiLU + split
__global__ void conv_silu_kernel(const float* __restrict__ zx,
                                 const float* __restrict__ cw,
                                 const float* __restrict__ cb,
                                 float* __restrict__ xb,
                                 float* __restrict__ Bm,
                                 float* __restrict__ Cm) {
    size_t idx = (size_t)blockIdx.x * blockDim.x + threadIdx.x;
    if (idx >= (size_t)ROWS * CONV_DIM) return;
    int c = (int)(idx % CONV_DIM);
    size_t bl = idx / CONV_DIM;
    int l = (int)(bl % LL);
    float acc = cb[c];
#pragma unroll
    for (int k = 0; k < D_CONV; k++) {
        int ls = l + k - (D_CONV - 1);
        if (ls >= 0)
            acc += zx[(bl + k - (D_CONV - 1)) * D_IN_PROJ + D_INNER + c] * cw[c * D_CONV + k];
    }
    float v = silu_f(acc);
    if (c < D_INNER)               xb[bl * D_INNER + c] = v;
    else if (c < D_INNER + D_STATE) Bm[bl * D_STATE + (c - D_INNER)] = v;
    else                            Cm[bl * D_STATE + (c - D_INNER - D_STATE)] = v;
}

// ------------------------------------------------------------ dt softplus + dA
__global__ void dt_kernel(const float* __restrict__ zx,
                          const float* __restrict__ dtb,
                          const float* __restrict__ alog,
                          float* __restrict__ dtv,
                          float* __restrict__ dAv) {
    int idx = blockIdx.x * blockDim.x + threadIdx.x;
    if (idx >= ROWS * NHEADS) return;
    int h = idx & (NHEADS - 1);
    size_t bl = (size_t)idx >> 4;
    float x = zx[bl * D_IN_PROJ + (D_IN_PROJ - NHEADS) + h] + dtb[h];
    float sp = (x > 20.f) ? x : log1pf(__expf(x));
    dtv[idx] = sp;
    dAv[idx] = __expf(-__expf(alog[h]) * sp);
}

// ------------------------------------------------------------ SSM scan
// One block per (b,h). 8 waves: wave w owns p = w*6 .. w*6+5; lane l owns
// state columns n = l and n+32 (registers). Per-step x/B/C broadcast via LDS,
// y[p] reduced within one wave32 via __shfl_xor.
__global__ __launch_bounds__(256)
void scan_kernel(const float* __restrict__ xb,
                 const float* __restrict__ Bm,
                 const float* __restrict__ Cm,
                 const float* __restrict__ dtv,
                 const float* __restrict__ dAv,
                 const float* __restrict__ Dp,
                 float* __restrict__ yb) {
    int bh = blockIdx.x;
    int b = bh >> 4, h = bh & (NHEADS - 1);
    int t = threadIdx.x;
    int w = t >> 5, l = t & 31;
    __shared__ float sx[HEADDIM], sB[D_STATE], sC[D_STATE];
    __shared__ float sdt, sdA;
    float s0[6], s1[6];
#pragma unroll
    for (int j = 0; j < 6; j++) { s0[j] = 0.f; s1[j] = 0.f; }
    float Dh = Dp[h];

    for (int step = 0; step < LL; step++) {
        size_t bl = (size_t)b * LL + step;
        if (t < 48)        sx[t]       = xb[(bl * NHEADS + h) * HEADDIM + t];
        else if (t < 112)  sB[t - 48]  = Bm[bl * D_STATE + (t - 48)];
        else if (t < 176)  sC[t - 112] = Cm[bl * D_STATE + (t - 112)];
        else if (t == 176) sdt = dtv[bl * NHEADS + h];
        else if (t == 177) sdA = dAv[bl * NHEADS + h];
        __syncthreads();
        float dt = sdt, dA = sdA;
        float bn0 = sB[l], bn1 = sB[l + 32];
        float cn0 = sC[l], cn1 = sC[l + 32];
#pragma unroll
        for (int j = 0; j < 6; j++) {
            int p = w * 6 + j;
            float xp = sx[p];
            float dx = dt * xp;
            s0[j] = s0[j] * dA + dx * bn0;
            s1[j] = s1[j] * dA + dx * bn1;
            float yp = s0[j] * cn0 + s1[j] * cn1;
#pragma unroll
            for (int off = 16; off > 0; off >>= 1)
                yp += __shfl_xor(yp, off, 32);
            if (l == 0)
                yb[(bl * NHEADS + h) * HEADDIM + p] = yp + Dh * xp;
        }
        __syncthreads();
    }
}

// ------------------------------------------------------------ gate (silu(z)) + RMSNorm -> bf16
__global__ void gate_rmsnorm_kernel(const float* __restrict__ yb,
                                    const float* __restrict__ zx,
                                    const float* __restrict__ gw,
                                    bf16_t* __restrict__ outbf) {
    int row = blockIdx.x, t = threadIdx.x;   // 256 threads, 3 cols each
    __shared__ float red[256];
    float u[3]; float ss = 0.f;
#pragma unroll
    for (int j = 0; j < 3; j++) {
        int c = t + j * 256;
        float z = zx[(size_t)row * D_IN_PROJ + c];
        float v = yb[(size_t)row * D_INNER + c] * silu_f(z);
        u[j] = v; ss += v * v;
    }
    red[t] = ss; __syncthreads();
    for (int st = 128; st > 0; st >>= 1) {
        if (t < st) red[t] += red[t + st];
        __syncthreads();
    }
    float sc = rsqrtf(red[0] * (1.f / D_INNER) + 1e-5f);
#pragma unroll
    for (int j = 0; j < 3; j++) {
        int c = t + j * 256;
        outbf[(size_t)row * D_INNER + c] = f2bf(u[j] * sc * gw[c]);
    }
}

// ------------------------------------------------------------ mean over L
__global__ void meanL_kernel(const float* __restrict__ hf, float* __restrict__ feat) {
    int b = blockIdx.x, d = threadIdx.x;   // 384 threads
    float s = 0.f;
    for (int l = 0; l < LL; l++)
        s += hf[((size_t)b * LL + l) * D_MODEL + d];
    feat[(size_t)b * D_MODEL + d] = s * (1.f / LL);
}

// ------------------------------------------------------------ classifier heads (HBM-bound)
__global__ void heads_kernel(const float* __restrict__ feat,
                             const float* __restrict__ w0, const float* __restrict__ b0,
                             const float* __restrict__ w1, const float* __restrict__ b1,
                             const float* __restrict__ w2, const float* __restrict__ b2,
                             const float* __restrict__ w3, const float* __restrict__ b3,
                             float* __restrict__ out) {
    __shared__ float sf[D_MODEL];
    int b = blockIdx.y;
    int o = blockIdx.x * 256 + threadIdx.x;
    for (int i = threadIdx.x; i < D_MODEL; i += 256)
        sf[i] = feat[(size_t)b * D_MODEL + i];
    __syncthreads();
    if (o >= N_OUT) return;
    const float* w; const float* bias; int col, n;
    if (o < N_ORDER)                       { w = w0; bias = b0; col = o; n = N_ORDER; }
    else if (o < N_ORDER + N_FAMILY)       { w = w1; bias = b1; col = o - N_ORDER; n = N_FAMILY; }
    else if (o < N_ORDER + N_FAMILY + N_GENUS)
                                           { w = w2; bias = b2; col = o - N_ORDER - N_FAMILY; n = N_GENUS; }
    else                                   { w = w3; bias = b3; col = o - N_ORDER - N_FAMILY - N_GENUS; n = N_SPECIES; }
    float acc = bias[col];
    for (int k = 0; k < D_MODEL; k++)
        acc += sf[k] * w[(size_t)k * n + col];
    out[(size_t)b * N_OUT + o] = acc;
}

// ================================================================ launch
extern "C" void kernel_launch(void* const* d_in, const int* in_sizes, int n_in,
                              void* d_out, int out_size, void* d_ws, size_t ws_size,
                              hipStream_t stream) {
    const int*   tokens     = (const int*)  d_in[0];
    const float* emb        = (const float*)d_in[1];
    const float* ln_w       = (const float*)d_in[2];
    const float* ln_b       = (const float*)d_in[3];
    const float* in_proj_w  = (const float*)d_in[4];
    const float* conv_w     = (const float*)d_in[5];
    const float* conv_b     = (const float*)d_in[6];
    const float* dt_bias    = (const float*)d_in[7];
    const float* A_log      = (const float*)d_in[8];
    const float* Dp         = (const float*)d_in[9];
    const float* gnorm_w    = (const float*)d_in[10];
    const float* out_proj_w = (const float*)d_in[11];
    const float* normf_w    = (const float*)d_in[12];
    const float* normf_b    = (const float*)d_in[13];
    const float* proj_ln_w  = (const float*)d_in[14];
    const float* proj_ln_b  = (const float*)d_in[15];
    const float* order_w    = (const float*)d_in[16];
    const float* order_b    = (const float*)d_in[17];
    const float* family_w   = (const float*)d_in[18];
    const float* family_b   = (const float*)d_in[19];
    const float* genus_w    = (const float*)d_in[20];
    const float* genus_b    = (const float*)d_in[21];
    const float* species_w  = (const float*)d_in[22];
    const float* species_b  = (const float*)d_in[23];
    float* out = (float*)d_out;

    // ---- workspace carve (bump allocator, 256B aligned) ----
    char* p = (char*)d_ws;
    auto alloc = [&](size_t bytes) -> void* {
        void* r = (void*)p;
        p += (bytes + 255) & ~(size_t)255;
        return r;
    };
    float*  resid   = (float*) alloc((size_t)ROWS * D_MODEL * 4);      // 25.2 MB
    bf16_t* hbf     = (bf16_t*)alloc((size_t)ROWS * D_MODEL * 2);      // 12.6 MB
    float*  zx      = (float*) alloc((size_t)ROWS * D_IN_PROJ * 4);    // 110 MB
    float*  xb      = (float*) alloc((size_t)ROWS * D_INNER * 4);      // 50.3 MB (reused as hf)
    float*  Bmb     = (float*) alloc((size_t)ROWS * D_STATE * 4);      // 4.2 MB
    float*  Cmb     = (float*) alloc((size_t)ROWS * D_STATE * 4);      // 4.2 MB
    float*  dtv     = (float*) alloc((size_t)ROWS * NHEADS * 4);       // 1 MB
    float*  dAv     = (float*) alloc((size_t)ROWS * NHEADS * 4);       // 1 MB
    float*  yb      = (float*) alloc((size_t)ROWS * D_INNER * 4);      // 50.3 MB
    bf16_t* gatedbf = (bf16_t*)alloc((size_t)ROWS * D_INNER * 2);      // 25.2 MB
    bf16_t* wInT    = (bf16_t*)alloc((size_t)D_IN_PROJ * D_MODEL * 2); // 1.3 MB
    bf16_t* wOutT   = (bf16_t*)alloc((size_t)D_MODEL * D_INNER * 2);   // 0.6 MB
    float*  featraw = (float*) alloc((size_t)BB * D_MODEL * 4);
    float*  feat    = (float*) alloc((size_t)BB * D_MODEL * 4);
    (void)ws_size; (void)in_sizes; (void)n_in; (void)out_size;

    // ---- 1. embedding gather ----
    {
        size_t n = (size_t)ROWS * D_MODEL;
        embed_kernel<<<(unsigned)((n + 255) / 256), 256, 0, stream>>>(tokens, emb, resid);
    }

    // ---- 2. mamba layers ----
    for (int l = 0; l < NLAYERS; l++) {
        // LN -> bf16 activations
        layernorm_kernel<<<ROWS, 128, 0, stream>>>(
            resid, ln_w + l * D_MODEL, ln_b + l * D_MODEL, hbf, nullptr);

        // in_proj weight -> bf16^T, then WMMA GEMM: zx = hbf @ Win
        {
            size_t n = (size_t)D_MODEL * D_IN_PROJ;
            convert_wT_kernel<<<(unsigned)((n + 255) / 256), 256, 0, stream>>>(
                in_proj_w + (size_t)l * D_MODEL * D_IN_PROJ, wInT, D_MODEL, D_IN_PROJ);
            int tilesN = D_IN_PROJ / 16;           // 105
            int total  = (ROWS / 64) * tilesN;     // 26880 wave-groups (64x16 each)
            wmma_gemm_bf16<<<total / 8, 256, 0, stream>>>(
                hbf, wInT, nullptr, zx, D_IN_PROJ, D_MODEL, tilesN, total);
        }

        // depthwise conv + SiLU + split
        {
            size_t n = (size_t)ROWS * CONV_DIM;
            conv_silu_kernel<<<(unsigned)((n + 255) / 256), 256, 0, stream>>>(
                zx, conv_w + (size_t)l * CONV_DIM * D_CONV, conv_b + (size_t)l * CONV_DIM,
                xb, Bmb, Cmb);
        }

        // dt softplus / dA
        dt_kernel<<<(ROWS * NHEADS + 255) / 256, 256, 0, stream>>>(
            zx, dt_bias + l * NHEADS, A_log + l * NHEADS, dtv, dAv);

        // sequential SSM scan (one block per (b,h))
        scan_kernel<<<BB * NHEADS, 256, 0, stream>>>(
            xb, Bmb, Cmb, dtv, dAv, Dp + l * NHEADS, yb);

        // gate with silu(z) + RMSNorm -> bf16
        gate_rmsnorm_kernel<<<ROWS, 256, 0, stream>>>(
            yb, zx, gnorm_w + l * D_INNER, gatedbf);

        // out_proj weight -> bf16^T, residual GEMM: resid += gated @ Wout
        {
            size_t n = (size_t)D_INNER * D_MODEL;
            convert_wT_kernel<<<(unsigned)((n + 255) / 256), 256, 0, stream>>>(
                out_proj_w + (size_t)l * D_INNER * D_MODEL, wOutT, D_INNER, D_MODEL);
            int tilesN = D_MODEL / 16;             // 24
            int total  = (ROWS / 64) * tilesN;     // 6144 wave-groups
            wmma_gemm_bf16<<<total / 8, 256, 0, stream>>>(
                gatedbf, wOutT, resid, resid, D_MODEL, D_INNER, tilesN, total);
        }
    }

    // ---- 3. final LN, mean over L, feature LN ----
    float* hf = xb;  // reuse (B,L,768) buffer for (B,L,384) fp32 LN output
    layernorm_kernel<<<ROWS, 128, 0, stream>>>(resid, normf_w, normf_b, nullptr, hf);
    meanL_kernel<<<BB, D_MODEL, 0, stream>>>(hf, featraw);
    layernorm_kernel<<<BB, 128, 0, stream>>>(featraw, proj_ln_w, proj_ln_b, nullptr, feat);

    // ---- 4. classifier heads (memory-bound) ----
    {
        dim3 g((N_OUT + 255) / 256, BB);
        heads_kernel<<<g, 256, 0, stream>>>(feat,
            order_w, order_b, family_w, family_b,
            genus_w, genus_b, species_w, species_b, out);
    }
}
